// ScaledDotProductAttention_32444182954172
// MI455X (gfx1250) — matrix operand order repaired and checked by reading, not compiled, not run on hardware
//
#include <hip/hip_runtime.h>

// ---------------------------------------------------------------------------
// Problem constants (from reference): B=2, H=16, S=2048, D=64, temp = 8
// d_out = output[B,H,S,D] (4,194,304 f32) ++ attn[B,H,S,S] (134,217,728 f32)
// ---------------------------------------------------------------------------
#define B_   2
#define H_   16
#define S_   2048
#define D_   64
#define BH_  (B_ * H_)
#define QKV_ELEMS (B_ * H_ * S_ * D_)      // 4,194,304
#define OUT_OFF   QKV_ELEMS                 // attn starts here in d_out
#define SCP  2052                           // padded LDS row stride (floats): m*2052 % 64 = m*4 -> conflict-free

typedef _Float16 v16h __attribute__((ext_vector_type(16)));
typedef _Float16 v8h  __attribute__((ext_vector_type(8)));
typedef _Float16 v4h  __attribute__((ext_vector_type(4)));
typedef float    v8f  __attribute__((ext_vector_type(8)));
typedef float    v4f  __attribute__((ext_vector_type(4)));

// ---------------------------------------------------------------------------
// Prep: f32 -> f16 with scale (q gets 1/temperature folded in; k gets 1.0)
// ---------------------------------------------------------------------------
__global__ __launch_bounds__(256) void cvt_f16_kernel(const float* __restrict__ src,
                                                      _Float16* __restrict__ dst,
                                                      float scale, int n4) {
  int i = blockIdx.x * 256 + threadIdx.x;
  if (i < n4) {
    v4f x = *(const v4f*)(src + (size_t)i * 4);
    v4h y;
    y[0] = (_Float16)(x[0] * scale);
    y[1] = (_Float16)(x[1] * scale);
    y[2] = (_Float16)(x[2] * scale);
    y[3] = (_Float16)(x[3] * scale);
    *(v4h*)(dst + (size_t)i * 4) = y;
  }
}

// ---------------------------------------------------------------------------
// Prep: v[bh][t][d] f32 -> vt[bh][d][t] f16  (tiled LDS transpose, coalesced
// both sides). Transposed V makes the WMMA B-fragment for P@V two contiguous
// 16B loads per lane.
// ---------------------------------------------------------------------------
__global__ __launch_bounds__(256) void transpose_v_kernel(const float* __restrict__ v,
                                                          _Float16* __restrict__ vt) {
  __shared__ float tile[64 * 65];
  const int bh = blockIdx.x >> 5;          // 32 t-blocks per bh
  const int t0 = (blockIdx.x & 31) << 6;   // 64 t-rows per tile
  const int tid = threadIdx.x;
  for (int i = tid; i < 64 * 64; i += 256) {
    int tl = i >> 6, d = i & 63;
    tile[tl * 65 + d] = v[((size_t)bh * S_ + t0 + tl) * D_ + d];
  }
  __syncthreads();
  for (int i = tid; i < 64 * 64; i += 256) {
    int d = i >> 6, tl = i & 63;
    vt[((size_t)bh * D_ + d) * S_ + t0 + tl] = (_Float16)tile[tl * 65 + d];
  }
}

// ---------------------------------------------------------------------------
// Main fused attention kernel.
// Grid: BH * (S/16) = 4096 blocks, 256 threads (8 wave32).
// Block = 16 query rows x 2048 keys for one (b,h).
//   phase 1: scores via v_wmma_f32_16x16x32_f16 (2 per 16x16 tile, K=64)
//            y = |s * exp(dwm/sigma^2)|   -> LDS strip [16][2052]
//   phase 2: rowwise softmax (max / exp / sum), float4 LDS traffic,
//            conflict-free bank mapping (thread c0 owns banks 4*c0..4*c0+3)
//   phase 3: write normalized attn to d_out, NON-TEMPORAL float4 stores
//            (537 MB streaming write must not thrash the 192 MB L2 that
//             caches q/k/v/dwm for all 4096 blocks)
//   phase 4: P@V via WMMA (A gathered from LDS f32->f16, B from transposed V),
//            4 waves x K-half each, LDS combine, scale by 1/rowsum, store out.
// ---------------------------------------------------------------------------
__global__ __launch_bounds__(256) void attn_fused_kernel(
    const _Float16* __restrict__ q16,  // [bh][s][d], already / temperature
    const _Float16* __restrict__ k16,  // [bh][t][d]
    const _Float16* __restrict__ vt16, // [bh][d][t]
    const float*    __restrict__ dwm,  // [b][s][t]
    const float*    __restrict__ sigma,
    float* __restrict__ out,           // [bh][s][d]
    float* __restrict__ attn)          // [bh][s][t]
{
  __shared__ float sc[16 * SCP];       // score / probability strip (131,328 B)
  __shared__ float rowmax[16];
  __shared__ float rinv[16];
  __shared__ float partial[4 * 256];   // PV partial sums

  const int tid     = threadIdx.x;
  const int wave    = tid >> 5;
  const int lane    = tid & 31;
  const int lane_lo = lane & 15;       // WMMA: N / M index within half-wave
  const int lane_hi = lane >> 4;       // WMMA: half-wave select

  const int bh = blockIdx.x >> 7;      // 128 row-blocks per (b,h)
  const int s0 = (blockIdx.x & 127) << 4;
  const int b  = bh >> 4;              // bh / H

  const float sg = sigma[0];
  const float inv_s2 = 1.0f / (sg * sg);

  const _Float16* qbase = q16  + ((size_t)bh * S_ + s0) * D_;
  const _Float16* kbase = k16  + (size_t)bh * S_ * D_;
  const _Float16* vbase = vt16 + (size_t)bh * D_ * S_;
  const float*    dwbase = dwm + ((size_t)b * S_ + s0) * S_;

  // --- A fragments for Q (row m = lane_lo; two K-chunks of 32) -------------
  // ISA A layout: v[0..3] hold K = 8*lane_hi + 0..7, v[4..7] hold K = +16.
  v16h aQ0, aQ1;
  {
    const _Float16* r = qbase + lane_lo * D_ + 8 * lane_hi;
    v8h x0 = *(const v8h*)(r);
    v8h x1 = *(const v8h*)(r + 16);
    v8h x2 = *(const v8h*)(r + 32);
    v8h x3 = *(const v8h*)(r + 48);
    #pragma unroll
    for (int i = 0; i < 8; ++i) {
      aQ0[i] = x0[i]; aQ0[8 + i] = x1[i];
      aQ1[i] = x2[i]; aQ1[8 + i] = x3[i];
    }
  }

  // --- phase 1: score tiles -------------------------------------------------
  for (int tt = wave; tt < 128; tt += 8) {
    const int t0 = tt << 4;
    // B fragment: lane column n = lane_lo, elements = K rows 8*lane_hi + e, +16 (same as A)
    // (k16 is [t][d] row-major, so Kt fragment elements are contiguous in d)
    const _Float16* kr = kbase + (size_t)(t0 + lane_lo) * D_ + 8 * lane_hi;
    v8h b0 = *(const v8h*)(kr);
    v8h b1 = *(const v8h*)(kr + 16);
    v8h b2 = *(const v8h*)(kr + 32);
    v8h b3 = *(const v8h*)(kr + 48);
    v16h bK0, bK1;
    #pragma unroll
    for (int i = 0; i < 8; ++i) {
      bK0[i] = b0[i]; bK0[8 + i] = b1[i];
      bK1[i] = b2[i]; bK1[8 + i] = b3[i];
    }
    v8f acc = {};
    acc = __builtin_amdgcn_wmma_f32_16x16x32_f16(false, aQ0, false, bK0,
                                                 (short)0, acc, false, false);
    acc = __builtin_amdgcn_wmma_f32_16x16x32_f16(false, aQ1, false, bK1,
                                                 (short)0, acc, false, false);
    const int n = t0 + lane_lo;
    #pragma unroll
    for (int r = 0; r < 8; ++r) {
      const int M = r + 8 * lane_hi;           // C/D layout: VGPR r -> row r+8*hi
      float g = __expf(dwbase[(size_t)M * S_ + n] * inv_s2);
      sc[M * SCP + n] = fabsf(acc[r] * g);
    }
  }
  __syncthreads();

  // --- phase 2: softmax stats (16 threads per row, float4 segments) --------
  {
    const int row = tid >> 4, c0 = tid & 15;
    const float* srow = &sc[row * SCP];
    float mx = -3.4e38f;
    for (int j = c0 * 4; j < S_; j += 64) {
      v4f x = *(const v4f*)&srow[j];
      mx = fmaxf(mx, fmaxf(fmaxf(x[0], x[1]), fmaxf(x[2], x[3])));
    }
    #pragma unroll
    for (int off = 8; off; off >>= 1) mx = fmaxf(mx, __shfl_xor(mx, off, 16));
    if (c0 == 0) rowmax[row] = mx;
  }
  __syncthreads();
  {
    const int row = tid >> 4, c0 = tid & 15;
    float* srow = &sc[row * SCP];
    const float mx = rowmax[row];
    float sm = 0.0f;
    for (int j = c0 * 4; j < S_; j += 64) {
      v4f x = *(const v4f*)&srow[j];
      v4f p;
      p[0] = __expf(x[0] - mx);
      p[1] = __expf(x[1] - mx);
      p[2] = __expf(x[2] - mx);
      p[3] = __expf(x[3] - mx);
      *(v4f*)&srow[j] = p;            // strip now holds unnormalized probs
      sm += p[0] + p[1] + p[2] + p[3];
    }
    #pragma unroll
    for (int off = 8; off; off >>= 1) sm += __shfl_xor(sm, off, 16);
    if (c0 == 0) rinv[row] = 1.0f / sm;
  }
  __syncthreads();

  // --- phase 3: write normalized attention (non-temporal float4) -----------
  {
    float* abase = attn + ((size_t)bh * S_ + s0) * S_;
    for (int i = tid; i < 16 * (S_ / 4); i += 256) {
      const int M = i >> 9;                   // 512 float4 per row
      const int t = (i & 511) << 2;
      v4f p = *(const v4f*)&sc[M * SCP + t];
      const float rv = rinv[M];
      p[0] *= rv; p[1] *= rv; p[2] *= rv; p[3] *= rv;
      __builtin_nontemporal_store(p, (v4f*)&abase[(size_t)M * S_ + t]);
    }
  }

  // --- phase 4: out = (P @ V) * rinv ---------------------------------------
  {
    const int ntile = wave & 3;               // d-column tile (4 x 16 = 64)
    const int khalf = wave >> 2;              // each half covers 1024 keys
    const int d0 = ntile << 4;
    v8f acc = {};
    const float*    prow = &sc[lane_lo * SCP];                  // A row m
    const _Float16* vrow = vbase + (size_t)(d0 + lane_lo) * S_; // B column n
    const int tbeg = khalf * 1024;
    for (int t0 = tbeg; t0 < tbeg + 1024; t0 += 32) {
      // A fragment from LDS probs (f32 -> f16 at load)
      v4f a0 = *(const v4f*)&prow[t0 + 8 * lane_hi];
      v4f a1 = *(const v4f*)&prow[t0 + 8 * lane_hi + 4];
      v4f a2 = *(const v4f*)&prow[t0 + 16 + 8 * lane_hi];
      v4f a3 = *(const v4f*)&prow[t0 + 16 + 8 * lane_hi + 4];
      v16h aP;
      #pragma unroll
      for (int i = 0; i < 4; ++i) {
        aP[i]      = (_Float16)a0[i];
        aP[4 + i]  = (_Float16)a1[i];
        aP[8 + i]  = (_Float16)a2[i];
        aP[12 + i] = (_Float16)a3[i];
      }
      // B fragment from transposed V: t = 8*lane_hi + 0..7 and +16 (same as A)
      const _Float16* vr = vrow + t0 + 8 * lane_hi;
      v8h b0 = *(const v8h*)(vr);
      v8h b1 = *(const v8h*)(vr + 16);
      v16h bV;
      #pragma unroll
      for (int i = 0; i < 8; ++i) { bV[i] = b0[i]; bV[8 + i] = b1[i]; }
      acc = __builtin_amdgcn_wmma_f32_16x16x32_f16(false, aP, false, bV,
                                                   (short)0, acc, false, false);
    }
    if (khalf == 1) *(v8f*)&partial[ntile * 256 + lane * 8] = acc;
    __syncthreads();
    if (khalf == 0) {
      v8f other = *(const v8f*)&partial[ntile * 256 + lane * 8];
      float* obase = out + ((size_t)bh * S_ + s0) * D_;
      #pragma unroll
      for (int r = 0; r < 8; ++r) {
        const int M = r + 8 * lane_hi;
        obase[(size_t)M * D_ + d0 + lane_lo] = (acc[r] + other[r]) * rinv[M];
      }
    }
  }
}

// ---------------------------------------------------------------------------
extern "C" void kernel_launch(void* const* d_in, const int* in_sizes, int n_in,
                              void* d_out, int out_size, void* d_ws, size_t ws_size,
                              hipStream_t stream) {
  const float* q     = (const float*)d_in[0];
  const float* k     = (const float*)d_in[1];
  const float* v     = (const float*)d_in[2];
  const float* dwm   = (const float*)d_in[3];
  const float* sigma = (const float*)d_in[4];

  float* out  = (float*)d_out;              // [B,H,S,D]
  float* attn = out + OUT_OFF;              // [B,H,S,S]

  _Float16* q16 = (_Float16*)d_ws;          // 8 MB
  _Float16* k16 = q16 + QKV_ELEMS;          // 8 MB
  _Float16* vt  = k16 + QKV_ELEMS;          // 8 MB (transposed)

  const int n4 = QKV_ELEMS / 4;             // 1,048,576 float4
  cvt_f16_kernel<<<(n4 + 255) / 256, 256, 0, stream>>>(q, q16, 0.125f, n4);
  cvt_f16_kernel<<<(n4 + 255) / 256, 256, 0, stream>>>(k, k16, 1.0f, n4);
  transpose_v_kernel<<<BH_ * (S_ / 64), 256, 0, stream>>>(v, vt);

  attn_fused_kernel<<<BH_ * (S_ / 16), 256, 0, stream>>>(
      q16, k16, vt, dwm, sigma, out, attn);
}
